// DOC_Retriever_49478023250050
// MI455X (gfx1250) — matrix operand order repaired and checked by reading, not compiled, not run on hardware
//
// MI455X (gfx1250) fused cosine-similarity retrieval.
//
// Roofline: feature matrix = 500000*768*4B = 1.536 GB, read exactly once
// -> 66us floor @ 23.3 TB/s HBM. Matmul FLOPs (147 GFLOP incl. 3-term split)
// are negligible at v_wmma_f32_16x16x32_bf16 rates, so the design is a pure
// memory stream with the norm, scaling, and top-k selection fused in.
//
// Numerics: fp32 values are split x = hi(bf16,trunc) + lo(bf16,rne) and the
// product is computed as qh*fh + qh*fl + ql*fh (error ~2^-17, fp32-class),
// so bf16 WMMA throughput with fp32-grade ranking fidelity.
//
// Query-norm division is dropped (positive row-constant, rank-invariant);
// feature norms are accumulated in registers during the streaming pass.
//
// CDNA5 specifics: WMMA fragments fed by ds_load_b128 from padded LDS;
// query chunks staged by the Tensor Data Mover (tensor_load_to_lds with
// pad_enable producing the conflict-free pitch directly), synchronized via
// s_wait_tensorcnt; feature stream warmed two chunks ahead with
// global_prefetch_b8.

#include <hip/hip_runtime.h>
#include <float.h>

typedef __bf16        bf16x16 __attribute__((ext_vector_type(16)));
typedef float         f32x8   __attribute__((ext_vector_type(8)));
typedef float         f32x4   __attribute__((ext_vector_type(4)));
typedef unsigned int  u32x4   __attribute__((ext_vector_type(4)));
typedef unsigned int  u32x8   __attribute__((ext_vector_type(8)));
typedef int           i32x4   __attribute__((ext_vector_type(4)));
typedef int           i32x8   __attribute__((ext_vector_type(8)));

#define BQ   64      // queries
#define DIM  768     // feature dim
#define NCH  12      // 768 / 64 K-chunks
#define FP   72      // LDS pitch (bf16 elems): 144B row stride -> conflict-free b128
#define TOPK 5
#define SEG  512

// LDS layout (byte offsets inside one block; static group segment starts at 0,
// which lets TDM descriptors carry absolute LDS addresses).
#define QBUF     9216u                 // one 64 x 72 bf16 stage
#define OFF_QH   0u                    // [2][64][FP] q hi (double buffered)
#define OFF_QL   18432u                // [2][64][FP] q lo
#define OFF_FH   36864u                // [64][FP]    f hi
#define OFF_FL   46080u                // [64][FP]    f lo
#define OFF_NORM 55296u                // [64][4] float
#define OFF_SIM  56320u                // [64][65] float
#define LDS_SZ   72960u

// ---------------- bf16 split helpers ----------------
static __device__ __forceinline__ unsigned pack_hi(float a, float b) {
  unsigned ua = __float_as_uint(a), ub = __float_as_uint(b);
  return (ua >> 16) | (ub & 0xFFFF0000u);            // truncated bf16 pair
}
static __device__ __forceinline__ float trunc_bf(float x) {
  return __uint_as_float(__float_as_uint(x) & 0xFFFF0000u);
}
static __device__ __forceinline__ unsigned pack_rne(float a, float b) {
  unsigned ua = __float_as_uint(a); ua += 0x7FFFu + ((ua >> 16) & 1u);
  unsigned ub = __float_as_uint(b); ub += 0x7FFFu + ((ub >> 16) & 1u);
  return (ua >> 16) | (ub & 0xFFFF0000u);            // RNE bf16 pair
}

// Build a 16xbf16 WMMA fragment from two 16B LDS reads (ds_load_b128 x2),
// elements 0..7 from p0, 8..15 from p1; callers use contiguous k = 16h + s for A and B.
static __device__ __forceinline__ bf16x16 ld_frag(const unsigned short* p0,
                                                  const unsigned short* p1) {
  u32x4 a = *(const u32x4*)p0;
  u32x4 b = *(const u32x4*)p1;
  u32x8 c;
  c[0] = a[0]; c[1] = a[1]; c[2] = a[2]; c[3] = a[3];
  c[4] = b[0]; c[5] = b[1]; c[6] = b[2]; c[7] = b[3];
  return __builtin_bit_cast(bf16x16, c);
}

// TDM: load a 64-row x 64-col bf16 tile (row stride DIM elems) into LDS with
// hardware padding: pad_interval = 32 DWORDs (64 bf16), pad_amount = 4 DWORDs
// (8 bf16) -> LDS row pitch 72 bf16 = FP. Descriptor per ISA 8.3/8.4:
//   g0: [1:0]=count=1, [63:32]=lds_addr, [120:64]=global_addr, [127:126]=type 2
//   g1: [17:16]=data_size(2B), [20]=pad_en, [24:22]=pad_interval, [31:25]=pad_amt,
//       [79:48]=tensor_dim0, [111:80]=tensor_dim1, [127:112]=tile_dim0,
//       [143:128]=tile_dim1, [207:160]=tensor_dim0_stride
// Groups 2/3 (3D+/gather) unused -> zeros. 6-arg builtin form (clang-23 lane).
static __device__ __forceinline__ void tdm_load_q(const unsigned short* gsrc,
                                                  unsigned ldsOff) {
  unsigned long long ga = (unsigned long long)(size_t)gsrc;
  u32x4 g0;
  g0[0] = 1u;                                        // count = 1 (user descriptor)
  g0[1] = ldsOff;                                    // LDS byte address
  g0[2] = (unsigned)ga;                              // global addr [31:0]
  g0[3] = (unsigned)(ga >> 32) | 0x80000000u;        // global addr MSBs | type=2
  i32x8 g1;
  g1[0] = (int)((1u << 16) | (1u << 20) | (4u << 22) | (3u << 25));
  g1[1] = (int)(((unsigned)DIM & 0xFFFFu) << 16);    // tensor_dim0 = 768 lo
  g1[2] = (int)(((unsigned)DIM >> 16) | (64u << 16));// tensor_dim0 hi | tensor_dim1=64
  g1[3] = (int)(64u << 16);                          // tile_dim0 = 64
  g1[4] = 64;                                        // tile_dim1 = 64
  g1[5] = DIM;                                       // tensor_dim0_stride = 768
  g1[6] = 0;
  g1[7] = 0;
  i32x4 z4 = {0, 0, 0, 0};
  i32x8 z8 = {0, 0, 0, 0, 0, 0, 0, 0};
  __builtin_amdgcn_tensor_load_to_lds(g0, g1, z4, z4, z8, 0);
}

// Sorted-descending top-K insert; tie -> smaller index (jax top_k semantics).
template <int K>
static __device__ __forceinline__ void topk_insert(float v, int id,
                                                   float (&val)[K], int (&idx)[K]) {
  bool ins = false;
#pragma unroll
  for (int j = 0; j < K; ++j) {
    bool better = (!ins) && ((v > val[j]) || (v == val[j] && id < idx[j]));
    if (better) {
#pragma unroll
      for (int t = K - 1; t > j; --t) { val[t] = val[t - 1]; idx[t] = idx[t - 1]; }
      val[j] = v; idx[j] = id;
      ins = true;
    }
  }
}

// ---------------- kernel 0: split queries into bf16 hi/lo ----------------
__global__ void qsplit_kernel(const float* __restrict__ q,
                              unsigned short* __restrict__ qhi,
                              unsigned short* __restrict__ qlo, int n) {
  int i = blockIdx.x * blockDim.x + threadIdx.x;
  if (i >= n) return;
  float x = q[i];
  float r = x - trunc_bf(x);                    // exact residual
  unsigned u = __float_as_uint(r); u += 0x7FFFu + ((u >> 16) & 1u);
  qhi[i] = (unsigned short)(__float_as_uint(x) >> 16);
  qlo[i] = (unsigned short)(u >> 16);
}

// ---------------- kernel 1: streamed GEMM + norms + per-WG top-5 ----------------
__global__ __launch_bounds__(256) void simtop_kernel(
    const float* __restrict__ feat, const unsigned short* __restrict__ qhiG,
    const unsigned short* __restrict__ qloG, float* __restrict__ cval,
    int* __restrict__ cidx, int N, int nTiles, int numWG) {
  __shared__ __align__(16) unsigned char sLDS[LDS_SZ];
  unsigned short* const sQh = (unsigned short*)(sLDS + OFF_QH);   // [2][64][FP]
  unsigned short* const sQl = (unsigned short*)(sLDS + OFF_QL);   // [2][64][FP]
  unsigned short* const sFh = (unsigned short*)(sLDS + OFF_FH);   // [64][FP]
  unsigned short* const sFl = (unsigned short*)(sLDS + OFF_FL);   // [64][FP]
  float* const sNorm = (float*)(sLDS + OFF_NORM);                 // [64][4]
  float* const sSim  = (float*)(sLDS + OFF_SIM);                  // [64][65]

  const int tid   = threadIdx.x;
  const int lane  = tid & 31;
  const int w     = tid >> 5;       // wave 0..7
  const int mw    = w & 3;          // 16-row M tile
  const int ng    = w >> 2;         // 32-col N group
  const int frow  = tid >> 2;       // loader row 0..63
  const int fq    = tid & 3;        // loader quarter (16 elems)
  const int lhalf = (lane < 16) ? 0 : 1;

  float tval[TOPK]; int tidx[TOPK];
#pragma unroll
  for (int j = 0; j < TOPK; ++j) { tval[j] = -FLT_MAX; tidx[j] = 0x7FFFFFFF; }

  const f32x4 Z4 = {0.f, 0.f, 0.f, 0.f};

  for (int tile = blockIdx.x; tile < nTiles; tile += numWG) {
    const int  n0    = tile * 64;
    const int  nRow  = n0 + frow;
    const bool rowOK = nRow < N;
    const float* fRow = feat + (size_t)(rowOK ? nRow : (N - 1)) * DIM + fq * 16;

    f32x8 acc0 = {}; f32x8 acc1 = {};
    float nacc = 0.f;

    // TDM prologue: wave 0 kicks q chunk 0 (hi+lo) into q buffer 0.
    if (w == 0) {
      tdm_load_q(qhiG, OFF_QH);
      tdm_load_q(qloG, OFF_QL);
    }

    // feature chunk 0 -> registers
    f32x4 bf0, bf1, bf2, bf3;
    if (rowOK) { bf0 = *(const f32x4*)fRow;       bf1 = *(const f32x4*)(fRow + 4);
                 bf2 = *(const f32x4*)(fRow + 8); bf3 = *(const f32x4*)(fRow + 12); }
    else       { bf0 = Z4; bf1 = Z4; bf2 = Z4; bf3 = Z4; }

    for (int c = 0; c < NCH; ++c) {
      const int qb = c & 1;
      __syncthreads();                 // previous chunk's LDS reads complete
      // ---- convert fp32 -> bf16 hi/lo, accumulate row norm, store to LDS ----
      {
        float x[16];
#pragma unroll
        for (int i = 0; i < 4; ++i) {
          x[i] = bf0[i]; x[4 + i] = bf1[i]; x[8 + i] = bf2[i]; x[12 + i] = bf3[i];
        }
        unsigned hu[8], lu[8];
#pragma unroll
        for (int i = 0; i < 8; ++i) {
          float a = x[2 * i], b = x[2 * i + 1];
          hu[i] = pack_hi(a, b);
          lu[i] = pack_rne(a - trunc_bf(a), b - trunc_bf(b));
          nacc += a * a + b * b;
        }
        unsigned short* fh = sFh + frow * FP + fq * 16;
        unsigned short* fl = sFl + frow * FP + fq * 16;
        u32x4 H0 = {hu[0], hu[1], hu[2], hu[3]}, H1 = {hu[4], hu[5], hu[6], hu[7]};
        u32x4 L0 = {lu[0], lu[1], lu[2], lu[3]}, L1 = {lu[4], lu[5], lu[6], lu[7]};
        *(u32x4*)fh = H0; *(u32x4*)(fh + 8) = H1;
        *(u32x4*)fl = L0; *(u32x4*)(fl + 8) = L1;
      }
      // ---- software-pipeline next feature chunk into registers ----
      if (c + 1 < NCH) {
        const float* p = fRow + (c + 1) * 64;
        if (rowOK) { bf0 = *(const f32x4*)p;       bf1 = *(const f32x4*)(p + 4);
                     bf2 = *(const f32x4*)(p + 8); bf3 = *(const f32x4*)(p + 12); }
        else       { bf0 = Z4; bf1 = Z4; bf2 = Z4; bf3 = Z4; }
      }
      if (c + 2 < NCH) {                     // warm L2 two chunks ahead
        __builtin_prefetch(fRow + (c + 2) * 64, 0, 3);
      }
      // ---- wave 0: complete TDM for chunk c, then kick chunk c+1 ----
      if (w == 0) {
        __builtin_amdgcn_s_wait_tensorcnt(0);
        if (c + 1 < NCH) {
          tdm_load_q(qhiG + (c + 1) * 64, OFF_QH + (unsigned)(qb ^ 1) * QBUF);
          tdm_load_q(qloG + (c + 1) * 64, OFF_QL + (unsigned)(qb ^ 1) * QBUF);
        }
      }
      __syncthreads();                 // chunk c (f via VALU, q via TDM) staged
      // ---- WMMA: two 32-wide K steps, 3-term bf16 split ----
#pragma unroll
      for (int k32 = 0; k32 < 64; k32 += 32) {
        const int arow = mw * 16 + (lane & 15);
        // A (16x32 bf16): lanes0-15 K0..15; lanes16-31 K16..31 (contiguous)
        const unsigned short* ah = sQh + qb * 4608 + arow * FP + k32 + lhalf * 16;
        const unsigned short* al = sQl + qb * 4608 + arow * FP + k32 + lhalf * 16;
        bf16x16 Ah = ld_frag(ah, ah + 8);
        bf16x16 Al = ld_frag(al, al + 8);
        // B (32x16 bf16): lanes0-15 K0..15; lanes16-31 K16..31 (contiguous)
        const int nc0 = ng * 32 + (lane & 15);
        const unsigned short* bh0 = sFh + nc0 * FP + k32 + lhalf * 16;
        const unsigned short* bl0 = sFl + nc0 * FP + k32 + lhalf * 16;
        bf16x16 Bh = ld_frag(bh0, bh0 + 8);
        bf16x16 Bl = ld_frag(bl0, bl0 + 8);
        acc0 = __builtin_amdgcn_wmma_f32_16x16x32_bf16(false, Ah, false, Bh, (short)0, acc0, false, false);
        acc0 = __builtin_amdgcn_wmma_f32_16x16x32_bf16(false, Ah, false, Bl, (short)0, acc0, false, false);
        acc0 = __builtin_amdgcn_wmma_f32_16x16x32_bf16(false, Al, false, Bh, (short)0, acc0, false, false);
        const unsigned short* bh1 = sFh + (nc0 + 16) * FP + k32 + lhalf * 16;
        const unsigned short* bl1 = sFl + (nc0 + 16) * FP + k32 + lhalf * 16;
        bf16x16 Bh1 = ld_frag(bh1, bh1 + 8);
        bf16x16 Bl1 = ld_frag(bl1, bl1 + 8);
        acc1 = __builtin_amdgcn_wmma_f32_16x16x32_bf16(false, Ah, false, Bh1, (short)0, acc1, false, false);
        acc1 = __builtin_amdgcn_wmma_f32_16x16x32_bf16(false, Ah, false, Bl1, (short)0, acc1, false, false);
        acc1 = __builtin_amdgcn_wmma_f32_16x16x32_bf16(false, Al, false, Bh1, (short)0, acc1, false, false);
      }
    }
    // ---- finalize norms, scale columns by rsqrt(|f|^2), stage sims ----
    sNorm[frow * 4 + fq] = nacc;
    __syncthreads();
#pragma unroll
    for (int t = 0; t < 2; ++t) {
      f32x8 a = t ? acc1 : acc0;
      int ncol = ng * 32 + t * 16 + (lane & 15);
      float nrm = sNorm[ncol * 4 + 0] + sNorm[ncol * 4 + 1] +
                  sNorm[ncol * 4 + 2] + sNorm[ncol * 4 + 3];
      float r = rsqrtf(fmaxf(nrm, 1e-30f));
      bool ok = (n0 + ncol) < N;
#pragma unroll
      for (int i = 0; i < 8; ++i) {          // C layout: VGPR i -> M=i (lanes<16) / 8+i
        int m = mw * 16 + lhalf * 8 + i;
        sSim[m * 65 + ncol] = ok ? a[i] * r : -FLT_MAX;
      }
    }
    __syncthreads();
    // ---- running top-5 per query row (threads 0..63, conflict-free reads) ----
    if (tid < 64) {
      for (int j = 0; j < 64; ++j) topk_insert(sSim[tid * 65 + j], n0 + j, tval, tidx);
    }
    __syncthreads();
  }

  if (tid < 64) {
    int base = blockIdx.x * (BQ * TOPK) + tid * TOPK;
#pragma unroll
    for (int j = 0; j < TOPK; ++j) { cval[base + j] = tval[j]; cidx[base + j] = tidx[j]; }
  }
}

// ---------------- kernel 2: merge candidates, gather data segments ----------------
__global__ __launch_bounds__(256) void topk_gather_kernel(
    const float* __restrict__ cval, const int* __restrict__ cidx,
    const int* __restrict__ data, int* __restrict__ out, int numWG, int N) {
  __shared__ float sv[256 * TOPK];
  __shared__ int   si[256 * TOPK];
  __shared__ int   fidx[TOPK];
  const int m = blockIdx.x, tid = threadIdx.x;

  float tval[TOPK]; int tidx[TOPK];
#pragma unroll
  for (int j = 0; j < TOPK; ++j) { tval[j] = -FLT_MAX; tidx[j] = 0x7FFFFFFF; }

  const int total = numWG * TOPK;
  for (int i = tid; i < total; i += 256) {
    int wg = i / TOPK, j = i - wg * TOPK;
    int off = wg * (BQ * TOPK) + m * TOPK + j;
    topk_insert(cval[off], cidx[off], tval, tidx);
  }
#pragma unroll
  for (int j = 0; j < TOPK; ++j) { sv[tid * TOPK + j] = tval[j]; si[tid * TOPK + j] = tidx[j]; }
  __syncthreads();
  if (tid == 0) {
    float fv[TOPK]; int fi[TOPK];
#pragma unroll
    for (int j = 0; j < TOPK; ++j) { fv[j] = -FLT_MAX; fi[j] = 0x7FFFFFFF; }
    for (int i = 0; i < 256 * TOPK; ++i) topk_insert(sv[i], si[i], fv, fi);
#pragma unroll
    for (int j = 0; j < TOPK; ++j) fidx[j] = fi[j];
  }
  __syncthreads();
#pragma unroll
  for (int j = 0; j < TOPK; ++j) {
    int doc = fidx[j];
    doc = doc < 0 ? 0 : (doc >= N ? N - 1 : doc);
    const i32x4* src = (const i32x4*)(data + (size_t)doc * SEG);
    i32x4* dst = (i32x4*)(out + ((size_t)m * TOPK + j) * SEG);
    for (int i = tid; i < SEG / 4; i += 256) dst[i] = src[i];
  }
}

// ---------------- host launcher ----------------
extern "C" void kernel_launch(void* const* d_in, const int* in_sizes, int n_in,
                              void* d_out, int out_size, void* d_ws, size_t ws_size,
                              hipStream_t stream) {
  (void)n_in; (void)out_size;
  const float* q    = (const float*)d_in[0];   // (64, 768) f32
  const float* feat = (const float*)d_in[1];   // (N, 768)  f32
  const int*   data = (const int*)d_in[2];     // (N, 512)  i32
  // d_in[3] is k (==5, fixed at compile time to match output shape)

  const int N      = in_sizes[1] / DIM;
  const int nTiles = (N + 63) / 64;

  int numWG = 2048;                            // candidate-list width, ws-capped
  while (numWG > 64 &&
         (size_t)(196608u) + (size_t)numWG * BQ * TOPK * 8u > ws_size)
    numWG >>= 1;
  if (numWG > nTiles) numWG = nTiles;
  if (numWG < 1) numWG = 1;

  unsigned short* qhi = (unsigned short*)d_ws;                 // 64*768 bf16
  unsigned short* qlo = qhi + BQ * DIM;                        // 64*768 bf16
  float* cval = (float*)((char*)d_ws + 196608);                // numWG*64*5 f32
  int*   cidx = (int*)(cval + (size_t)numWG * BQ * TOPK);      // numWG*64*5 i32

  qsplit_kernel<<<(BQ * DIM + 255) / 256, 256, 0, stream>>>(q, qhi, qlo, BQ * DIM);
  simtop_kernel<<<numWG, 256, 0, stream>>>(feat, qhi, qlo, cval, cidx, N, nTiles, numWG);
  topk_gather_kernel<<<64, 256, 0, stream>>>(cval, cidx, data, (int*)d_out, numWG, N);
}
